// FastestFlashMHA_738734374969
// MI455X (gfx1250) — matrix operand order repaired and checked by reading, not compiled, not run on hardware
//
#include <hip/hip_runtime.h>
#include <math.h>

// ---------------- problem constants ----------------
constexpr int Bb   = 2;
constexpr int Sdim = 2048;
constexpr int HID  = 2048;
constexpr int NH   = 16;
constexpr int HD   = 128;
constexpr int Mtot = Bb * Sdim;          // 4096 rows in the big GEMMs
constexpr int WINDOW = 1024;
constexpr float SCALEF  = 0.08838834764831845f;   // 1/sqrt(128)
constexpr float SOFTCAP = 30.0f;

// GEMM tiling: 128x128 block tile, 8 waves, each wave 16 rows x 128 cols
constexpr int BM = 128, BN = 128, BK = 32;

// ---------------- vector types ----------------
typedef __attribute__((ext_vector_type(16))) __bf16          v16bf;
typedef __attribute__((ext_vector_type(8)))  float           v8f;
typedef __attribute__((ext_vector_type(8)))  unsigned short  u16x8;
typedef __attribute__((ext_vector_type(16))) unsigned short  u16x16;
typedef __attribute__((ext_vector_type(4)))  float           f32x4;

static __device__ __forceinline__ unsigned short f2bfu(float f) {
    unsigned u = __builtin_bit_cast(unsigned, f);
    u += 0x7FFFu + ((u >> 16) & 1u);            // round-to-nearest-even
    return (unsigned short)(u >> 16);
}

static __device__ __forceinline__ v16bf make16(u16x8 lo, u16x8 hi) {
    union { u16x16 u; v16bf v; } cv;
    cv.u = __builtin_shufflevector(lo, hi, 0,1,2,3,4,5,6,7,8,9,10,11,12,13,14,15);
    return cv.v;
}

static __device__ __forceinline__ v16bf as16(u16x16 u) {
    union { u16x16 u; v16bf v; } cv; cv.u = u; return cv.v;
}

static __device__ __forceinline__ v8f wmma_bf16(v16bf a, v16bf b, v8f c) {
    // (neg_a, A, neg_b, B, c_mod, C, reuse_a, reuse_b)
    return __builtin_amdgcn_wmma_f32_16x16x32_bf16(false, a, false, b,
                                                   (short)0, c, false, false);
}

// =====================================================================
// GEMM: Y = X @ W^T   (X: [Mtot,HID], W: [HID,HID] row-major, K-contig)
// MODE 0: store bf16 as [B,H,S,D]      (Q, K)
// MODE 1: store bf16 as [B,H,D,S]      (V transposed)
// MODE 2: store fp32 as [Mtot,HID]     (final output)
// BF16_IN: X is bf16 (ushort) instead of fp32
// =====================================================================
template<int MODE, bool BF16_IN>
__global__ __launch_bounds__(256) void gemm_xwt(const void* __restrict__ Xv,
                                                const float* __restrict__ W,
                                                void* __restrict__ Yv)
{
    __shared__ alignas(32) unsigned short Xs[BM * BK];   // 128x32 bf16 (8KB)
    __shared__ alignas(32) unsigned short Ws[BN * BK];   // 128x32 bf16 (8KB)

    const int tid  = threadIdx.x;
    const int lane = tid & 31;
    const int wid  = tid >> 5;          // 0..7  -> 16-row slice of the block tile
    const int g    = lane >> 4;         // half-wave group
    const int ln   = lane & 15;
    const int m0   = blockIdx.y * BM;
    const int n0   = blockIdx.x * BN;

    // staging assignment: each thread owns 16 contiguous K elems of one row
    const int sr = tid >> 1;            // 0..127
    const int sc = (tid & 1) * 16;      // 0 or 16

    v8f acc[8] = {};

    for (int k0 = 0; k0 < HID; k0 += BK) {
        // ---- prefetch next k-tile (global_prefetch_b8) ----
        if (k0 + BK < HID) {
            if (BF16_IN)
                __builtin_prefetch((const unsigned short*)Xv +
                                   (size_t)(m0 + sr) * HID + k0 + BK + sc, 0, 3);
            else
                __builtin_prefetch((const float*)Xv +
                                   (size_t)(m0 + sr) * HID + k0 + BK + sc, 0, 3);
            __builtin_prefetch(W + (size_t)(n0 + sr) * HID + k0 + BK + sc, 0, 3);
        }

        // ---- stage X tile (128x32 bf16), 16 elems / thread ----
        if (BF16_IN) {
            const unsigned short* X = (const unsigned short*)Xv;
            u16x16 v = *(const u16x16*)(X + (size_t)(m0 + sr) * HID + k0 + sc);
            *(u16x16*)(Xs + sr * BK + sc) = v;
        } else {
            const float* X = (const float*)Xv;
            const f32x4* p = (const f32x4*)(X + (size_t)(m0 + sr) * HID + k0 + sc);
            f32x4 q0 = p[0], q1 = p[1], q2 = p[2], q3 = p[3];
            u16x16 v;
            #pragma unroll
            for (int i = 0; i < 4; i++) {
                v[i]      = f2bfu(q0[i]);
                v[4 + i]  = f2bfu(q1[i]);
                v[8 + i]  = f2bfu(q2[i]);
                v[12 + i] = f2bfu(q3[i]);
            }
            *(u16x16*)(Xs + sr * BK + sc) = v;
        }
        // ---- stage W tile (128x32 bf16), 16 elems / thread ----
        {
            const f32x4* p = (const f32x4*)(W + (size_t)(n0 + sr) * HID + k0 + sc);
            f32x4 q0 = p[0], q1 = p[1], q2 = p[2], q3 = p[3];
            u16x16 v;
            #pragma unroll
            for (int i = 0; i < 4; i++) {
                v[i]      = f2bfu(q0[i]);
                v[4 + i]  = f2bfu(q1[i]);
                v[8 + i]  = f2bfu(q2[i]);
                v[12 + i] = f2bfu(q3[i]);
            }
            *(u16x16*)(Ws + sr * BK + sc) = v;
        }
        __syncthreads();

        // ---- one A tile per wave, reused across 8 WMMAs ----
        const unsigned short* xa = Xs + (wid * 16 + ln) * BK;
        v16bf a = make16(*(const u16x8*)(xa + 16 * g),
                         *(const u16x8*)(xa + 16 * g + 8));

        #pragma unroll
        for (int nt = 0; nt < 8; nt++) {
            u16x16 bu = *(const u16x16*)(Ws + (nt * 16 + ln) * BK + 16 * g);
            acc[nt] = wmma_bf16(a, as16(bu), acc[nt]);
        }
        __syncthreads();
    }

    // ---- epilogue ----
    #pragma unroll
    for (int nt = 0; nt < 8; nt++) {
        #pragma unroll
        for (int r = 0; r < 8; r++) {
            const int row = m0 + wid * 16 + r + 8 * g;
            const int col = n0 + nt * 16 + ln;
            const float val = acc[nt][r];
            if (MODE == 2) {
                ((float*)Yv)[(size_t)row * HID + col] = val;
            } else {
                const int b = row >> 11;        // / Sdim
                const int s = row & (Sdim - 1);
                const int h = col >> 7;         // / HD
                const int d = col & (HD - 1);
                unsigned short* Y = (unsigned short*)Yv;
                if (MODE == 0)
                    Y[(((size_t)(b * NH + h)) * Sdim + s) * HD + d] = f2bfu(val);
                else
                    Y[(((size_t)(b * NH + h)) * HD + d) * Sdim + s] = f2bfu(val);
            }
        }
    }
}

// =====================================================================
// Flash attention with softcap + causal sliding window.
// One wave (32 threads) per 16-row query block. Q:[B,H,S,D] K:[B,H,S,D]
// Vt:[B,H,D,S] all bf16. Output attn: [B,S,H*D] bf16.
// =====================================================================
__global__ __launch_bounds__(32) void flash_attn(const unsigned short* __restrict__ Q,
                                                 const unsigned short* __restrict__ Kk,
                                                 const unsigned short* __restrict__ Vt,
                                                 unsigned short* __restrict__ A)
{
    __shared__ alignas(32) unsigned short pLds[16 * 32];  // P tile bf16

    const int lane = threadIdx.x & 31;
    const int g    = lane >> 4;
    const int ln   = lane & 15;
    const int bh   = blockIdx.x >> 7;        // / (S/16 = 128)
    const int qb   = blockIdx.x & 127;
    const int i0   = qb * 16;
    const int b    = bh >> 4;
    const int h    = bh & (NH - 1);

    const unsigned short* Qb = Q  + (size_t)bh * Sdim * HD;
    const unsigned short* Kb = Kk + (size_t)bh * Sdim * HD;
    const unsigned short* Vb = Vt + (size_t)bh * HD * Sdim;

    // ---- Q tile (16x128) in registers, A-layout, 4 k-steps of 32 ----
    v16bf aq[4];
    const int qrow = i0 + ln;
    #pragma unroll
    for (int ks = 0; ks < 4; ks++) {
        const unsigned short* p = Qb + (size_t)qrow * HD + ks * 32;
        aq[ks] = make16(*(const u16x8*)(p + 16 * g),
                        *(const u16x8*)(p + 16 * g + 8));
    }

    v8f acc[8];
    float mrow[8], lrow[8];
    #pragma unroll
    for (int r = 0; r < 8; r++) { acc[r] = (v8f){}; mrow[r] = -1e30f; lrow[r] = 0.f; }

    int tstart = i0 - (WINDOW - 1);
    if (tstart < 0) tstart = 0;
    tstart &= ~31;
    const int tend = i0 + 16;

    for (int t0 = tstart; t0 < tend; t0 += 32) {
        // ---- prefetch next key block's K rows ----
        if (t0 + 32 < tend) {
            __builtin_prefetch(Kb + (size_t)(t0 + 32 + ln) * HD + 16 * g, 0, 3);
            __builtin_prefetch(Kb + (size_t)(t0 + 48 + ln) * HD + 16 * g, 0, 3);
        }

        // ---- S = Q K^T for two 16-col tiles ----
        v8f c0 = {}, c1 = {};
        #pragma unroll
        for (int ks = 0; ks < 4; ks++) {
            const unsigned short* kp0 = Kb + (size_t)(t0 + ln)      * HD + ks * 32 + 16 * g;
            const unsigned short* kp1 = Kb + (size_t)(t0 + 16 + ln) * HD + ks * 32 + 16 * g;
            u16x16 b0 = *(const u16x16*)kp0;
            u16x16 b1 = *(const u16x16*)kp1;
            c0 = wmma_bf16(aq[ks], as16(b0), c0);
            c1 = wmma_bf16(aq[ks], as16(b1), c1);
        }

        // ---- scale, softcap, mask ----
        float p0[8], p1[8], mx[8];
        #pragma unroll
        for (int r = 0; r < 8; r++) {
            const int irow = i0 + r + 8 * g;
            float s0 = SOFTCAP * tanhf(c0[r] * (SCALEF / SOFTCAP));
            float s1 = SOFTCAP * tanhf(c1[r] * (SCALEF / SOFTCAP));
            const int ta = t0 + ln, tb = t0 + 16 + ln;
            const bool ok0 = (ta <= irow) && (irow - ta < WINDOW);
            const bool ok1 = (tb <= irow) && (irow - tb < WINDOW);
            p0[r] = ok0 ? s0 : -1e30f;
            p1[r] = ok1 ? s1 : -1e30f;
            mx[r] = fmaxf(p0[r], p1[r]);
        }
        // ---- row max across the 16-lane N-group ----
        #pragma unroll
        for (int r = 0; r < 8; r++) {
            #pragma unroll
            for (int msk = 1; msk < 16; msk <<= 1)
                mx[r] = fmaxf(mx[r], __shfl_xor(mx[r], msk, 32));
        }
        float corr[8], rsum[8];
        #pragma unroll
        for (int r = 0; r < 8; r++) {
            const float mnew = fmaxf(mrow[r], mx[r]);
            corr[r] = __expf(mrow[r] - mnew);
            mrow[r] = mnew;
            const float e0 = (p0[r] > -1e29f) ? __expf(p0[r] - mnew) : 0.f;
            const float e1 = (p1[r] > -1e29f) ? __expf(p1[r] - mnew) : 0.f;
            p0[r] = e0; p1[r] = e1;
            rsum[r] = e0 + e1;
        }
        #pragma unroll
        for (int r = 0; r < 8; r++) {
            #pragma unroll
            for (int msk = 1; msk < 16; msk <<= 1)
                rsum[r] += __shfl_xor(rsum[r], msk, 32);
            lrow[r] = lrow[r] * corr[r] + rsum[r];
        }

        // ---- P: C-layout -> A-layout via LDS ----
        #pragma unroll
        for (int r = 0; r < 8; r++) {
            pLds[(r + 8 * g) * 32 + ln]      = f2bfu(p0[r]);
            pLds[(r + 8 * g) * 32 + 16 + ln] = f2bfu(p1[r]);
        }
        __syncthreads();
        const unsigned short* pp = pLds + ln * 32;
        v16bf pa = make16(*(const u16x8*)(pp + 16 * g),
                          *(const u16x8*)(pp + 16 * g + 8));

        // ---- O = O*corr + P V ----
        #pragma unroll
        for (int dt = 0; dt < 8; dt++) {
            const unsigned short* vp = Vb + (size_t)(dt * 16 + ln) * Sdim + t0 + 16 * g;
            u16x16 bv = *(const u16x16*)vp;
            v8f cs;
            #pragma unroll
            for (int r = 0; r < 8; r++) cs[r] = acc[dt][r] * corr[r];
            acc[dt] = wmma_bf16(pa, as16(bv), cs);
        }
        __syncthreads();
    }

    // ---- epilogue: attn[b][i][h*HD+d] (bf16) ----
    float rinv[8];
    #pragma unroll
    for (int r = 0; r < 8; r++) rinv[r] = __frcp_rn(lrow[r]);
    #pragma unroll
    for (int dt = 0; dt < 8; dt++) {
        #pragma unroll
        for (int r = 0; r < 8; r++) {
            const int irow = i0 + r + 8 * g;
            const int d    = dt * 16 + ln;
            const float v  = acc[dt][r] * rinv[r];
            A[((size_t)b * Sdim + irow) * HID + h * HD + d] = f2bfu(v);
        }
    }
}

// =====================================================================
extern "C" void kernel_launch(void* const* d_in, const int* in_sizes, int n_in,
                              void* d_out, int out_size, void* d_ws, size_t ws_size,
                              hipStream_t stream)
{
    const float* x  = (const float*)d_in[0];
    const float* Wq = (const float*)d_in[1];
    const float* Wk = (const float*)d_in[2];
    const float* Wv = (const float*)d_in[3];
    const float* Wo = (const float*)d_in[4];
    float* out = (float*)d_out;

    const size_t nElem = (size_t)Mtot * HID;      // 8M bf16 per buffer
    unsigned short* Qw  = (unsigned short*)d_ws;
    unsigned short* Kw  = Qw  + nElem;
    unsigned short* Vtw = Kw  + nElem;
    unsigned short* Aw  = Vtw + nElem;

    dim3 gGemm(HID / BN, Mtot / BM);   // (16, 32)
    dim3 bGemm(256);

    gemm_xwt<0, false><<<gGemm, bGemm, 0, stream>>>((const void*)x, Wq, (void*)Qw);
    gemm_xwt<0, false><<<gGemm, bGemm, 0, stream>>>((const void*)x, Wk, (void*)Kw);
    gemm_xwt<1, false><<<gGemm, bGemm, 0, stream>>>((const void*)x, Wv, (void*)Vtw);

    flash_attn<<<Bb * NH * (Sdim / 16), 32, 0, stream>>>(Qw, Kw, Vtw, Aw);

    gemm_xwt<2, true><<<gGemm, bGemm, 0, stream>>>((const void*)Aw, Wo, (void*)out);
}
